// LSTM_30159260352545
// MI455X (gfx1250) — matrix operand order repaired and checked by reading, not compiled, not run on hardware
//
#include <hip/hip_runtime.h>
#include <hip/hip_bf16.h>

// ---------------------------------------------------------------------------
// LSTM on MI455X (gfx1250), wave32 + v_wmma_f32_16x16x32_bf16.
//  Phase 1: xg[s][b][g] = x@Wx^T + bh   (big bf16 WMMA GEMM, software-pipelined)
//  Phase 2: 1024 sequential step kernels; each workgroup TDM-stages its
//           256 KB Wh slice into LDS (tensor_load_to_lds + s_wait_tensorcnt),
//           runs a pipelined WMMA loop from LDS, and does lane-local gate math
//           with xg/c operands prefetched under the GEMM.
// ---------------------------------------------------------------------------

typedef __attribute__((ext_vector_type(16))) __bf16 v16bf;
typedef __attribute__((ext_vector_type(8)))  __bf16 v8bf;
typedef __attribute__((ext_vector_type(8)))  float  v8f;
typedef __attribute__((ext_vector_type(4)))  unsigned int v4u;
typedef __attribute__((ext_vector_type(8)))  int v8i;
typedef __attribute__((ext_vector_type(4)))  int v4i;

#define B_   32
#define S_   1024
#define I_   512
#define H_   512
#define G4_  2048
#define BSH_ (32u * 1024u * 512u)

// workspace layout (bytes)
#define XG_OFF   0ull          // fp32 [S][B][4H]            256 MB
#define XBF_OFF  268435456ull  // bf16 x [B][S][I]            32 MB
#define WXBF_OFF 301989888ull  // bf16 Wx [4H][I]              2 MB
#define WHBF_OFF 304087040ull  // bf16 Wh [4H][H]              2 MB
#define HBF0_OFF 306184192ull  // bf16 h buf0 [B][H]          32 KB
#define HBF1_OFF 306216960ull  // bf16 h buf1 [B][H]          32 KB
#define C_OFF    306249728ull  // fp32 c [B][H]               64 KB
#define BH_OFF   306315264ull  // fp32 bh_all [4H]             8 KB

static __device__ __forceinline__ unsigned short f2bf(float f) {
  unsigned int u = __float_as_uint(f);
  u += 0x7FFFu + ((u >> 16) & 1u);       // round-to-nearest-even
  return (unsigned short)(u >> 16);
}
static __device__ __forceinline__ float sigm(float x) {
  return 1.0f / (1.0f + __expf(-x));
}

union U16 { v16bf v; v8bf h[2]; };

// --------------------------- phase 0: convert/init -------------------------
__global__ __launch_bounds__(256) void lstm_convert_init(
    const float* __restrict__ x,
    const float* __restrict__ Wxf, const float* __restrict__ Wxi,
    const float* __restrict__ Wxo, const float* __restrict__ Wxc,
    const float* __restrict__ Whf, const float* __restrict__ Whi,
    const float* __restrict__ Who, const float* __restrict__ Whc,
    const float* __restrict__ bhf, const float* __restrict__ bhi,
    const float* __restrict__ bho, const float* __restrict__ bhc,
    const float* __restrict__ c0,  const float* __restrict__ h0,
    unsigned short* __restrict__ xbf,  unsigned short* __restrict__ wxbf,
    unsigned short* __restrict__ whbf, unsigned short* __restrict__ hbf0,
    float* __restrict__ c, float* __restrict__ bh_all)
{
  unsigned t = blockIdx.x * 256u + threadIdx.x;
  if (t < (unsigned)(B_ * S_ * I_)) xbf[t] = f2bf(x[t]);
  if (t < (unsigned)(G4_ * I_)) {
    unsigned gate = t >> 18, rem = t & 262143u;  // 512*512 per gate
    const float* wx = gate == 0 ? Wxf : gate == 1 ? Wxi : gate == 2 ? Wxo : Wxc;
    const float* wh = gate == 0 ? Whf : gate == 1 ? Whi : gate == 2 ? Who : Whc;
    wxbf[t] = f2bf(wx[rem]);
    whbf[t] = f2bf(wh[rem]);
  }
  if (t < (unsigned)G4_) {
    unsigned gate = t >> 9;
    const float* bp = gate == 0 ? bhf : gate == 1 ? bhi : gate == 2 ? bho : bhc;
    bh_all[t] = bp[t & 511u];
  }
  if (t < (unsigned)(B_ * H_)) {
    c[t]    = c0[t & 511u];
    hbf0[t] = f2bf(h0[t & 511u]);
  }
}

// ------------------- phase 1: xg = x @ Wx^T + bh (WMMA) ---------------------
// Wave task: one 16-row M-tile x four consecutive 16-col N-tiles, K=512.
// Software-pipelined: prefetch tile k+32 while WMMAs consume tile k.
__global__ __launch_bounds__(256) void lstm_xproj(
    const unsigned short* __restrict__ xbf,
    const unsigned short* __restrict__ wxbf,
    const float* __restrict__ bh_all,
    float* __restrict__ xg)
{
  const int lane  = threadIdx.x & 31;
  const int task  = blockIdx.x * 8 + (threadIdx.x >> 5);
  const int mtile = task >> 5;          // 0..2047  (rows of [S*B, I])
  const int ng    = task & 31;          // group of 4 N-tiles (64 cols)
  const int col16 = lane & 15;
  const int hi    = (lane & 16) ? 1 : 0;
  const int sel8  = hi ? 8 : 0;         // unused: A is loaded with selB, lanes 16-31 hold K 16-31
  const int selB  = hi ? 16 : 0;        // B: lanes 16-31 hold K 16-31

  // A row r maps to (s = r/32, b = r%32); x is [B][S][I]
  const int r = mtile * 16 + col16;
  const unsigned short* aptr =
      xbf + ((size_t)(r & 31) << 19) + ((size_t)(r >> 5) << 9);

  const unsigned short* bptr[4];
#pragma unroll
  for (int nn = 0; nn < 4; ++nn) {
    int nrow = ng * 64 + nn * 16 + col16;   // B column n = Wx row n
    bptr[nn] = wxbf + (size_t)nrow * 512 + selB;
  }

  v8f acc[4] = {};

  // prologue: load tile k=0
  U16 a;
  v16bf bm[4];
  a.h[0] = *(const v8bf*)(aptr + selB);
  a.h[1] = *(const v8bf*)(aptr + selB + 8);
#pragma unroll
  for (int nn = 0; nn < 4; ++nn) bm[nn] = *(const v16bf*)(bptr[nn]);

#pragma unroll 3
  for (int k0 = 0; k0 < 480; k0 += 32) {
    // prefetch tile k0+32 into the shadow set
    U16 an;
    v16bf bn[4];
    const int kn = k0 + 32;
    an.h[0] = *(const v8bf*)(aptr + kn + selB);
    an.h[1] = *(const v8bf*)(aptr + kn + selB + 8);
#pragma unroll
    for (int nn = 0; nn < 4; ++nn) bn[nn] = *(const v16bf*)(bptr[nn] + kn);
    // consume tile k0
#pragma unroll
    for (int nn = 0; nn < 4; ++nn)
      acc[nn] = __builtin_amdgcn_wmma_f32_16x16x32_bf16(
          false, a.v, false, bm[nn], (short)0, acc[nn], false, false);
    a = an;
#pragma unroll
    for (int nn = 0; nn < 4; ++nn) bm[nn] = bn[nn];
  }
  // epilogue: last tile (k=480)
#pragma unroll
  for (int nn = 0; nn < 4; ++nn)
    acc[nn] = __builtin_amdgcn_wmma_f32_16x16x32_bf16(
        false, a.v, false, bm[nn], (short)0, acc[nn], false, false);

  const int rbase = hi ? 8 : 0;
#pragma unroll
  for (int nn = 0; nn < 4; ++nn) {
    int n = ng * 64 + nn * 16 + col16;
    float bias = bh_all[n];
#pragma unroll
    for (int j = 0; j < 8; ++j) {
      int rr = mtile * 16 + rbase + j;       // rr = s*32 + b, matches [S][B]
      xg[(size_t)rr * 2048 + n] = acc[nn][j] + bias;
    }
  }
}

// ----------------- phase 2: one recurrent step (TDM + WMMA + gates) ---------
// 8 blocks x 8 waves. Block handles hct0..hct0+3 for BOTH batch m-tiles, so
// its Wh slice = 4 gates x 64 rows x 512 k bf16 = 256 KB -> fits in WGP LDS.
// Wave w: mtile = w&1, hct = hct0 + (w>>1).
__global__ __launch_bounds__(256) void lstm_step(
    const float* __restrict__ xg,
    const unsigned short* __restrict__ whbf,
    const unsigned short* __restrict__ hread,
    unsigned short* __restrict__ hwrite,
    float* __restrict__ c,
    float* __restrict__ out_o,
    float* __restrict__ out_h,
    int s)
{
  extern __shared__ unsigned short smem[];   // 4 * 64 * 512 bf16 = 256 KB

  const int lane  = threadIdx.x & 31;
  const int w     = threadIdx.x >> 5;   // wave 0..7
  const int hct0  = blockIdx.x * 4;
  const int mtile = w & 1;              // batch tile 0..1
  const int hctl  = w >> 1;             // local h-col tile 0..3
  const int hct   = hct0 + hctl;
  const int col16 = lane & 15;
  const int hi    = (lane & 16) ? 1 : 0;
  const int sel8  = hi ? 8 : 0;
  const int selB  = hi ? 16 : 0;
  const int rbase = hi ? 8 : 0;
  const int hidx  = hct * 16 + col16;

  // --- prefetch epilogue operands (independent of the GEMM) -----------------
  float xv[4][8];
  float cv[8];
#pragma unroll
  for (int j = 0; j < 8; ++j) {
    int b = mtile * 16 + rbase + j;
    size_t xb = ((size_t)s * 32 + (size_t)b) * 2048;
#pragma unroll
    for (int g = 0; g < 4; ++g) xv[g][j] = xg[xb + (size_t)(g * 512) + hidx];
    cv[j] = c[(size_t)b * 512 + hidx];
  }

  // --- TDM: stage 4 gate slices of Wh (each 64 rows x 512 bf16) into LDS ---
  if (w == 0) {
#pragma unroll
    for (int g = 0; g < 4; ++g) {
      unsigned long long gaddr =
          (unsigned long long)(whbf + (size_t)(g * 512 + hct0 * 16) * 512);
      v4u g0;
      g0[0] = 1u;                                   // count=1, normal load D#
      g0[1] = (unsigned)(g * 65536);                // lds_addr (bytes)
      g0[2] = (unsigned)(gaddr & 0xFFFFFFFFull);    // global_addr[31:0]
      g0[3] = (unsigned)((gaddr >> 32) & 0x01FFFFFFull) | 0x80000000u; // +type=2
      v8i g1;
      g1[0] = (int)(1u << 16);     // workgroup_mask=0, data_size=1 (2 bytes)
      g1[1] = (int)(512u << 16);   // tensor_dim0 = 512 (low 16 bits)
      g1[2] = (int)(64u << 16);    // tensor_dim0 hi=0 | tensor_dim1 = 64
      g1[3] = (int)(512u << 16);   // tensor_dim1 hi=0 | tile_dim0 = 512
      g1[4] = 64;                  // tile_dim1 = 64, tile_dim2 = 0
      g1[5] = 512;                 // tensor_dim0_stride[31:0] = 512
      g1[6] = 0;                   // stride hi / dim1_stride lo
      g1[7] = 0;                   // dim1_stride hi
      v4i z4 = {0, 0, 0, 0};       // 2-D tensor: groups 2/3 unused
      v8i z8 = {0, 0, 0, 0, 0, 0, 0, 0};
      __builtin_amdgcn_tensor_load_to_lds(g0, g1, z4, z4, z8, 0);
    }
    __builtin_amdgcn_s_wait_tensorcnt(0);
  }
  __syncthreads();

  const unsigned short* aptr = hread + (size_t)(mtile * 16 + col16) * 512;
  const unsigned short* bptr[4];
#pragma unroll
  for (int g = 0; g < 4; ++g) {
    int lrow = g * 64 + hctl * 16 + col16;   // local row inside LDS slice
    bptr[g] = smem + (size_t)lrow * 512 + selB;
  }

  v8f acc[4] = {};

  // prologue: load tile k=0 (A from global h, B from LDS)
  U16 a;
  U16 bm[4];
  a.h[0] = *(const v8bf*)(aptr + selB);
  a.h[1] = *(const v8bf*)(aptr + selB + 8);
#pragma unroll
  for (int g = 0; g < 4; ++g) {
    bm[g].h[0] = *(const v8bf*)(bptr[g]);
    bm[g].h[1] = *(const v8bf*)(bptr[g] + 8);
  }

#pragma unroll 3
  for (int k0 = 0; k0 < 480; k0 += 32) {
    U16 an;
    U16 bn[4];
    const int kn = k0 + 32;
    an.h[0] = *(const v8bf*)(aptr + kn + selB);
    an.h[1] = *(const v8bf*)(aptr + kn + selB + 8);
#pragma unroll
    for (int g = 0; g < 4; ++g) {
      bn[g].h[0] = *(const v8bf*)(bptr[g] + kn);       // ds_load_b128
      bn[g].h[1] = *(const v8bf*)(bptr[g] + kn + 8);   // ds_load_b128
    }
#pragma unroll
    for (int g = 0; g < 4; ++g)
      acc[g] = __builtin_amdgcn_wmma_f32_16x16x32_bf16(
          false, a.v, false, bm[g].v, (short)0, acc[g], false, false);
    a = an;
#pragma unroll
    for (int g = 0; g < 4; ++g) bm[g] = bn[g];
  }
#pragma unroll
  for (int g = 0; g < 4; ++g)
    acc[g] = __builtin_amdgcn_wmma_f32_16x16x32_bf16(
        false, a.v, false, bm[g].v, (short)0, acc[g], false, false);

  // --- gate math: everything lane-local, operands already in registers ------
#pragma unroll
  for (int j = 0; j < 8; ++j) {
    int b = mtile * 16 + rbase + j;
    float f  = sigm(acc[0][j] + xv[0][j]);
    float i  = sigm(acc[1][j] + xv[1][j]);
    float o  = sigm(acc[2][j] + xv[2][j]);
    float ch = sigm(acc[3][j] + xv[3][j]);   // sigmoid per reference
    size_t ci = (size_t)b * 512 + hidx;
    float cn = f * cv[j] + i * ch;
    c[ci] = cn;
    float hn = o * tanhf(cn);
    hwrite[ci] = f2bf(hn);
    size_t oi = (size_t)b * (S_ * H_) + (size_t)s * H_ + (size_t)hidx;
    out_o[oi] = o;
    out_h[oi] = hn;
  }
}

// --------------------------------- launch ----------------------------------
extern "C" void kernel_launch(void* const* d_in, const int* in_sizes, int n_in,
                              void* d_out, int out_size, void* d_ws, size_t ws_size,
                              hipStream_t stream) {
  (void)in_sizes; (void)n_in; (void)out_size; (void)ws_size;
  char* ws = (char*)d_ws;
  float*          xg     = (float*)(ws + XG_OFF);
  unsigned short* xbf    = (unsigned short*)(ws + XBF_OFF);
  unsigned short* wxbf   = (unsigned short*)(ws + WXBF_OFF);
  unsigned short* whbf   = (unsigned short*)(ws + WHBF_OFF);
  unsigned short* hbf0   = (unsigned short*)(ws + HBF0_OFF);
  unsigned short* hbf1   = (unsigned short*)(ws + HBF1_OFF);
  float*          cbuf   = (float*)(ws + C_OFF);
  float*          bh_all = (float*)(ws + BH_OFF);

  const float* x   = (const float*)d_in[0];
  const float* Wxf = (const float*)d_in[1];
  const float* Wxi = (const float*)d_in[2];
  const float* Wxo = (const float*)d_in[3];
  const float* Wxc = (const float*)d_in[4];
  const float* Whf = (const float*)d_in[5];
  const float* Whi = (const float*)d_in[6];
  const float* Who = (const float*)d_in[7];
  const float* Whc = (const float*)d_in[8];
  const float* bhf = (const float*)d_in[9];
  const float* bhi = (const float*)d_in[10];
  const float* bho = (const float*)d_in[11];
  const float* bhc = (const float*)d_in[12];
  const float* c0  = (const float*)d_in[13];
  const float* h0  = (const float*)d_in[14];

  // phase 0: bf16 packing + state init (covers 16.7M elements)
  lstm_convert_init<<<65536, 256, 0, stream>>>(
      x, Wxf, Wxi, Wxo, Wxc, Whf, Whi, Who, Whc,
      bhf, bhi, bho, bhc, c0, h0,
      xbf, wxbf, whbf, hbf0, cbuf, bh_all);

  // phase 1: big input-projection GEMM. 65536 wave tiles, 8 waves/block.
  lstm_xproj<<<8192, 256, 0, stream>>>(xbf, wxbf, bh_all, xg);

  // phase 2: sequential recurrence, one kernel per step (graph = 1026 nodes)
  float* out_o = (float*)d_out;
  float* out_h = out_o + (size_t)BSH_;
  for (int s = 0; s < S_; ++s) {
    const unsigned short* hr = (s & 1) ? hbf1 : hbf0;
    unsigned short*       hw = (s & 1) ? hbf0 : hbf1;   // double-buffered h
    lstm_step<<<8, 256, 262144, stream>>>(xg, whbf, hr, hw, cbuf,
                                          out_o, out_h, s);
  }
}
